// BipartiteRGAT_27049704030449
// MI455X (gfx1250) — matrix operand order repaired and checked by reading, not compiled, not run on hardware
//
#include <hip/hip_runtime.h>
#include <hip/hip_bf16.h>

// ---------------------------------------------------------------------------
// BipartiteRGAT on MI455X (gfx1250, wave32).
// GEMMs: v_wmma_f32_16x16x32_bf16, branch-free inner loop, all operands
// pre-converted to bf16 with K padded to a multiple of 32 so every fragment
// slice is one aligned global_load_b128. Each wave computes a 16x64 strip
// (4 WMMAs share one A fragment). Edge softmax/aggregation is scatter/atomic
// bound -> VALU + global atomics.
// ---------------------------------------------------------------------------

typedef __attribute__((ext_vector_type(16))) __bf16 v16bf;
typedef __attribute__((ext_vector_type(8)))  __bf16 v8bf;
typedef __attribute__((ext_vector_type(8)))  float  v8f;

#define NEG_SLOPE 0.2f

__device__ __forceinline__ unsigned short f32_to_bf16_rne(float f) {
  union { float f; unsigned u; } cv; cv.f = f;
  unsigned u = cv.u;
  u += 0x7FFFu + ((u >> 16) & 1u);     // round to nearest even
  return (unsigned short)(u >> 16);
}

__device__ __forceinline__ __bf16 bf16_of(float f) {
  union { unsigned short s; __bf16 b; } cv;
  cv.s = f32_to_bf16_rne(f);
  return cv.b;
}

__device__ __forceinline__ void atomicMaxF32(float* addr, float val) {
  unsigned* ua = (unsigned*)addr;
  unsigned old = *ua;
  for (;;) {
    float cur = __uint_as_float(old);
    if (cur >= val) break;
    unsigned assumed = old;
    old = atomicCAS(ua, assumed, __float_as_uint(val));
    if (old == assumed) break;
  }
}

// ---------------------------------------------------------------------------
// f32 [rows x K] -> bf16 [rows x Kp] (zero padded).  Bandwidth bound.
// ---------------------------------------------------------------------------
__global__ void cvt_pad_bf16(const float* __restrict__ in, __bf16* __restrict__ out,
                             int rows, int K, int Kp) {
  int i = blockIdx.x * blockDim.x + threadIdx.x;
  if (i >= rows * Kp) return;
  int r = i / Kp, k = i - r * Kp;
  out[i] = (k < K) ? bf16_of(in[(size_t)r * K + k]) : bf16_of(0.f);
}

// W [R x K x Nout] f32 -> Wt [R x Nout x K] bf16 (transpose-convert; tiny)
__global__ void wtrans_bf16(const float* __restrict__ W, __bf16* __restrict__ Wt,
                            int R, int K, int Nout) {
  int i = blockIdx.x * blockDim.x + threadIdx.x;
  if (i >= R * K * Nout) return;
  int r = i / (K * Nout);
  int rem = i - r * K * Nout;
  int n = rem / K, k = rem - n * K;
  Wt[i] = bf16_of(W[((size_t)r * K + k) * Nout + n]);
}

// ---------------------------------------------------------------------------
// C[M x Ncols](f32) = act(A_bf16[M x Kp] @ Bt_bf16[Ncols x Kp]^T + bias)
// Per-relation strides via blockIdx.z. One wave -> 16x64 strip (4 WMMAs/kstep).
// Fragment layouts (A/B: this code's contiguous k order, same in both; C/D per ISA):
//  A (bf16 16x32): lane -> row = lane&15 ; 16 contiguous elems @ k0+16*laneHi
//  B (bf16 32x16): lane -> col = lane&15 ; 16 contiguous elems @ k0+16*laneHi
//  C/D (f32 16x16): lane -> col = lane&15 ; vgpr v -> row = v + 8*laneHi
// ---------------------------------------------------------------------------
__global__ void gemm_bf16(const __bf16* __restrict__ A,
                          const __bf16* __restrict__ Bt,
                          const float* __restrict__ bias,
                          float* __restrict__ Cout,
                          int M, int Kp, int Ncols, int ldc,
                          unsigned long long bt_stride,
                          unsigned long long out_stride,
                          int relu) {
  const __bf16* B = Bt + (size_t)blockIdx.z * bt_stride;
  float* C = Cout + (size_t)blockIdx.z * out_stride;

  const int wave   = threadIdx.x >> 5;
  const int lane   = threadIdx.x & 31;
  const int laneHi = lane >> 4;
  const int l16    = lane & 15;

  const int groups_n = Ncols >> 6;          // 64-wide strips
  const int tiles    = (M >> 4) * groups_n;
  const int tile = blockIdx.x * (blockDim.x >> 5) + wave;
  if (tile >= tiles) return;
  const int tm = tile / groups_n;
  const int tg = tile - tm * groups_n;
  const int row0 = tm * 16;
  const int col0 = tg * 64;

  // per-lane base pointers (all 16B aligned: Kp % 32 == 0)
  const __bf16* aptr = A + (size_t)(row0 + l16) * Kp + 16 * laneHi;
  const __bf16* bp0  = B + (size_t)(col0 + 0  + l16) * Kp + 16 * laneHi;
  const __bf16* bp1  = B + (size_t)(col0 + 16 + l16) * Kp + 16 * laneHi;
  const __bf16* bp2  = B + (size_t)(col0 + 32 + l16) * Kp + 16 * laneHi;
  const __bf16* bp3  = B + (size_t)(col0 + 48 + l16) * Kp + 16 * laneHi;

  v8f c0 = {}, c1 = {}, c2 = {}, c3 = {};
  union F16 { v16bf v; v8bf h[2]; };

  for (int k0 = 0; k0 < Kp; k0 += 32) {
    F16 a, b0, b1, b2, b3;
    a.h[0]  = *(const v8bf*)(aptr + k0);
    a.h[1]  = *(const v8bf*)(aptr + k0 + 8);
    b0.h[0] = *(const v8bf*)(bp0 + k0);
    b0.h[1] = *(const v8bf*)(bp0 + k0 + 8);
    b1.h[0] = *(const v8bf*)(bp1 + k0);
    b1.h[1] = *(const v8bf*)(bp1 + k0 + 8);
    b2.h[0] = *(const v8bf*)(bp2 + k0);
    b2.h[1] = *(const v8bf*)(bp2 + k0 + 8);
    b3.h[0] = *(const v8bf*)(bp3 + k0);
    b3.h[1] = *(const v8bf*)(bp3 + k0 + 8);
    c0 = __builtin_amdgcn_wmma_f32_16x16x32_bf16(false, a.v, false, b0.v, (short)0, c0, false, false);
    c1 = __builtin_amdgcn_wmma_f32_16x16x32_bf16(false, a.v, false, b1.v, (short)0, c1, false, false);
    c2 = __builtin_amdgcn_wmma_f32_16x16x32_bf16(false, a.v, false, b2.v, (short)0, c2, false, false);
    c3 = __builtin_amdgcn_wmma_f32_16x16x32_bf16(false, a.v, false, b3.v, (short)0, c3, false, false);
  }

  float bb0 = bias ? bias[col0 + 0  + l16] : 0.f;
  float bb1 = bias ? bias[col0 + 16 + l16] : 0.f;
  float bb2 = bias ? bias[col0 + 32 + l16] : 0.f;
  float bb3 = bias ? bias[col0 + 48 + l16] : 0.f;
#pragma unroll
  for (int v = 0; v < 8; ++v) {
    int mm = v + 8 * laneHi;
    float* crow = C + (size_t)(row0 + mm) * ldc + col0 + l16;
    float v0 = c0[v] + bb0, v1 = c1[v] + bb1, v2 = c2[v] + bb2, v3 = c3[v] + bb3;
    if (relu) {
      v0 = v0 > 0.f ? v0 : 0.f;  v1 = v1 > 0.f ? v1 : 0.f;
      v2 = v2 > 0.f ? v2 : 0.f;  v3 = v3 > 0.f ? v3 : 0.f;
    }
    crow[0]  = v0;
    crow[16] = v1;
    crow[32] = v2;
    crow[48] = v3;
  }
}

__global__ void fill_f32(float* __restrict__ p, float v, int n) {
  int i = blockIdx.x * blockDim.x + threadIdx.x;
  if (i < n) p[i] = v;
}

// alpha = leaky_relu(qi + kj); running segment max into mbuf[dst][h]
__global__ void edge_alpha_kernel(const float* __restrict__ xw,
                                  const int* __restrict__ src,
                                  const int* __restrict__ dst,
                                  const int* __restrict__ et,
                                  const float* __restrict__ q,
                                  const float* __restrict__ kk,
                                  float* __restrict__ alpha,
                                  float* __restrict__ mbuf,
                                  int E, int heads, int HC, int Nnodes) {
  __shared__ float sq[512];
  __shared__ float sk[512];
  const int qn = HC * heads;
  for (int i = threadIdx.x; i < qn; i += blockDim.x) { sq[i] = q[i]; sk[i] = kk[i]; }
  __syncthreads();

  int e = blockIdx.x * blockDim.x + threadIdx.x;
  if (e >= E) return;
  int s = src[e], d = dst[e], r = et[e];
  const float* oi = xw + ((size_t)r * Nnodes + d) * HC;   // dest / query
  const float* oj = xw + ((size_t)r * Nnodes + s) * HC;   // source / key
  float accq[4] = {0.f, 0.f, 0.f, 0.f};
  float acck[4] = {0.f, 0.f, 0.f, 0.f};
  for (int o = 0; o < HC; ++o) {
    float vi = oi[o], vj = oj[o];
    for (int h = 0; h < heads; ++h) {
      accq[h] += vi * sq[o * heads + h];
      acck[h] += vj * sk[o * heads + h];
    }
  }
  for (int h = 0; h < heads; ++h) {
    float al = accq[h] + acck[h];
    al = al > 0.f ? al : NEG_SLOPE * al;
    alpha[(size_t)e * heads + h] = al;
    atomicMaxF32(&mbuf[(size_t)d * heads + h], al);
  }
}

// a = exp(alpha - m[dst]); segment-sum into sbuf[dst][h]
__global__ void edge_expsum_kernel(const int* __restrict__ dst,
                                   float* __restrict__ alpha,
                                   const float* __restrict__ mbuf,
                                   float* __restrict__ sbuf,
                                   int E, int heads) {
  int i = blockIdx.x * blockDim.x + threadIdx.x;
  if (i >= E * heads) return;
  int e = i / heads, h = i - e * heads;
  int d = dst[e];
  float a = __expf(alpha[i] - mbuf[(size_t)d * heads + h]);
  alpha[i] = a;
  atomicAdd(&sbuf[(size_t)d * heads + h], a);
}

// agg[dst][c] += (a / (s[dst][h] + eps)) * xw[etype][src][c]
__global__ void edge_agg_kernel(const float* __restrict__ xw,
                                const int* __restrict__ src,
                                const int* __restrict__ dst,
                                const int* __restrict__ et,
                                const float* __restrict__ alpha,
                                const float* __restrict__ sbuf,
                                float* __restrict__ agg,
                                int E, int heads, int C, int HC, int Nnodes) {
  long long i = (long long)blockIdx.x * blockDim.x + threadIdx.x;
  long long total = (long long)E * HC;
  if (i >= total) return;
  int e = (int)(i / HC);
  int c = (int)(i - (long long)e * HC);
  int h = c / C;
  int d = dst[e];
  float a = alpha[(size_t)e * heads + h] /
            (sbuf[(size_t)d * heads + h] + 1e-16f);
  float v = xw[((size_t)et[e] * Nnodes + src[e]) * HC + c];
  atomicAdd(&agg[(size_t)d * HC + c], a * v);
}

// out_f32 = relu(in + bias) ; out_bf16 = bf16(out_f32)  (next layer's A matrix)
__global__ void bias_relu_cvt_kernel(const float* __restrict__ in,
                                     const float* __restrict__ bias,
                                     float* __restrict__ outf,
                                     __bf16* __restrict__ outb,
                                     int n, int HC) {
  int i = blockIdx.x * blockDim.x + threadIdx.x;
  if (i >= n) return;
  float v = in[i] + bias[i % HC];
  v = v > 0.f ? v : 0.f;
  outf[i] = v;
  outb[i] = bf16_of(v);
}

// out[n][l] = x[n] . Wl[l] + bl[l]  (L = 5: pure VALU)
__global__ void classifier_kernel(const float* __restrict__ x,
                                  const float* __restrict__ Wl,
                                  const float* __restrict__ bl,
                                  float* __restrict__ out,
                                  int M, int C, int L) {
  int i = blockIdx.x * blockDim.x + threadIdx.x;
  if (i >= M * L) return;
  int n = i / L, l = i - n * L;
  const float* xr = x + (size_t)n * C;
  const float* wr = Wl + (size_t)l * C;
  float acc = bl[l];
  for (int c = 0; c < C; ++c) acc += xr[c] * wr[c];
  out[i] = acc;
}

// ---------------------------------------------------------------------------
extern "C" void kernel_launch(void* const* d_in, const int* in_sizes, int n_in,
                              void* d_out, int out_size, void* d_ws, size_t ws_size,
                              hipStream_t stream) {
  const float* x0  = (const float*)d_in[0];
  const float* x1  = (const float*)d_in[1];
  const float* x2  = (const float*)d_in[2];
  const int* eidx  = (const int*)d_in[3];
  const int* etype = (const int*)d_in[4];
  const float* Wp0 = (const float*)d_in[5];
  const float* bp0 = (const float*)d_in[6];
  const float* Wp1 = (const float*)d_in[7];
  const float* bp1 = (const float*)d_in[8];
  const float* Wp2 = (const float*)d_in[9];
  const float* bp2 = (const float*)d_in[10];
  const float* W1  = (const float*)d_in[11];
  const float* q1  = (const float*)d_in[12];
  const float* k1  = (const float*)d_in[13];
  const float* b1  = (const float*)d_in[14];
  const float* W2  = (const float*)d_in[15];
  const float* q2  = (const float*)d_in[16];
  const float* k2  = (const float*)d_in[17];
  const float* b2  = (const float*)d_in[18];
  const float* Wl  = (const float*)d_in[19];
  const float* bl  = (const float*)d_in[20];

  const int E = in_sizes[4];          // 400000
  const int* src = eidx;
  const int* dst = eidx + E;

  const int N = 30000, HC = 128;
  // padded K sizes (multiples of 32)
  const int K0p = 2016, K1p = 1504, K2p = 1024;

  // ---- workspace layout (bytes, all 16B aligned) --------------------------
  char* ws = (char*)d_ws;
  float*  x_nodes = (float*)(ws + 0);             // 30000*256 f32   (30.72 MB)
  __bf16* xb      = (__bf16*)(ws + 30720000);     // 30000*256 bf16  (15.36 MB)
  __bf16* Wp0b    = (__bf16*)(ws + 46080000);     // 256*2016 bf16
  __bf16* Wp1b    = (__bf16*)(ws + 47112192);     // 256*1504 bf16
  __bf16* Wp2b    = (__bf16*)(ws + 47882240);     // 256*1024 bf16
  __bf16* W1t     = (__bf16*)(ws + 48406528);     // 6*128*256 bf16
  __bf16* W2t     = (__bf16*)(ws + 48799744);     // 6*128*128 bf16
  // union region @ 49,000,000: {x0b,x1b,x2b} (projection stage) / xw (layers)
  __bf16* x0b     = (__bf16*)(ws + 49000000);     // 10000*2016 bf16 (40.32 MB)
  __bf16* x1b     = (__bf16*)(ws + 89320000);     // 10000*1504 bf16 (30.08 MB)
  __bf16* x2b     = (__bf16*)(ws + 119400000);    // 10000*1024 bf16 (20.48 MB)
  float*  xw      = (float*)(ws + 49000000);      // 6*30000*128 f32 (92.16 MB)
  float*  agg     = (float*)(ws + 141200000);     // 30000*128 f32   (15.36 MB)
  float*  alpha   = (float*)(ws + 156560000);     // 400000*4 f32    ( 6.40 MB)
  float*  mbuf    = (float*)(ws + 162960000);     // 30000*4 f32
  float*  sbuf    = (float*)(ws + 163440000);     // 30000*4 f32  -> ends ~163.9 MB

  dim3 blk(256);
  auto cdiv = [](long long a, long long b) { return (int)((a + b - 1) / b); };

  // ---- stage bf16 operands ------------------------------------------------
  cvt_pad_bf16<<<cdiv((long long)10000 * K0p, 256), blk, 0, stream>>>(x0, x0b, 10000, 2000, K0p);
  cvt_pad_bf16<<<cdiv((long long)10000 * K1p, 256), blk, 0, stream>>>(x1, x1b, 10000, 1500, K1p);
  cvt_pad_bf16<<<cdiv((long long)10000 * K2p, 256), blk, 0, stream>>>(x2, x2b, 10000, 1000, K2p);
  cvt_pad_bf16<<<cdiv(256 * K0p, 256), blk, 0, stream>>>(Wp0, Wp0b, 256, 2000, K0p);
  cvt_pad_bf16<<<cdiv(256 * K1p, 256), blk, 0, stream>>>(Wp1, Wp1b, 256, 1500, K1p);
  cvt_pad_bf16<<<cdiv(256 * K2p, 256), blk, 0, stream>>>(Wp2, Wp2b, 256, 1000, K2p);
  wtrans_bf16<<<cdiv(6 * 256 * 128, 256), blk, 0, stream>>>(W1, W1t, 6, 256, 128);
  wtrans_bf16<<<cdiv(6 * 128 * 128, 256), blk, 0, stream>>>(W2, W2t, 6, 128, 128);

  // ---- projections: x_nodes = relu(x_i @ Wp_i^T + bp_i) -------------------
  // waves = (10000/16) * (256/64) = 2500 -> 313 blocks (8 waves/block)
  gemm_bf16<<<313, blk, 0, stream>>>(x0b, Wp0b, bp0, x_nodes,                     10000, K0p, 256, 256, 0, 0, 1);
  gemm_bf16<<<313, blk, 0, stream>>>(x1b, Wp1b, bp1, x_nodes + (size_t)10000*256, 10000, K1p, 256, 256, 0, 0, 1);
  gemm_bf16<<<313, blk, 0, stream>>>(x2b, Wp2b, bp2, x_nodes + (size_t)20000*256, 10000, K2p, 256, 256, 0, 0, 1);

  // bf16 copy of node features for layer-1 A matrix
  cvt_pad_bf16<<<cdiv((long long)N * 256, 256), blk, 0, stream>>>(x_nodes, xb, N, 256, 256);

  // ---- layer 1 (heads=4, C1=32, HC=128, K=256) ----------------------------
  // waves/relation = 1875 * (128/64) = 3750 -> 469 blocks, gridDim.z = 6
  gemm_bf16<<<dim3(469, 1, 6), blk, 0, stream>>>(xb, W1t, nullptr, xw, N, 256, 128, 128,
                                                 (unsigned long long)128 * 256,
                                                 (unsigned long long)N * 128, 0);
  fill_f32<<<cdiv(N * 4, 256),   blk, 0, stream>>>(mbuf, -1e30f, N * 4);
  fill_f32<<<cdiv(N * 4, 256),   blk, 0, stream>>>(sbuf, 0.f,    N * 4);
  fill_f32<<<cdiv(N * 128, 256), blk, 0, stream>>>(agg,  0.f,    N * 128);
  edge_alpha_kernel<<<cdiv(E, 256), blk, 0, stream>>>(xw, src, dst, etype, q1, k1, alpha, mbuf, E, 4, HC, N);
  edge_expsum_kernel<<<cdiv((long long)E * 4, 256), blk, 0, stream>>>(dst, alpha, mbuf, sbuf, E, 4);
  edge_agg_kernel<<<cdiv((long long)E * HC, 256), blk, 0, stream>>>(xw, src, dst, etype, alpha, sbuf, agg, E, 4, 32, HC, N);
  bias_relu_cvt_kernel<<<cdiv(N * HC, 256), blk, 0, stream>>>(agg, b1, x_nodes, xb, N * HC, HC);

  // ---- layer 2 (heads=1, C2=128, HC=128, K=128) ---------------------------
  gemm_bf16<<<dim3(469, 1, 6), blk, 0, stream>>>(xb, W2t, nullptr, xw, N, 128, 128, 128,
                                                 (unsigned long long)128 * 128,
                                                 (unsigned long long)N * 128, 0);
  fill_f32<<<cdiv(N, 256),       blk, 0, stream>>>(mbuf, -1e30f, N);
  fill_f32<<<cdiv(N, 256),       blk, 0, stream>>>(sbuf, 0.f,    N);
  fill_f32<<<cdiv(N * 128, 256), blk, 0, stream>>>(agg,  0.f,    N * 128);
  edge_alpha_kernel<<<cdiv(E, 256), blk, 0, stream>>>(xw, src, dst, etype, q2, k2, alpha, mbuf, E, 1, HC, N);
  edge_expsum_kernel<<<cdiv(E, 256), blk, 0, stream>>>(dst, alpha, mbuf, sbuf, E, 1);
  edge_agg_kernel<<<cdiv((long long)E * HC, 256), blk, 0, stream>>>(xw, src, dst, etype, alpha, sbuf, agg, E, 1, 128, HC, N);
  bias_relu_cvt_kernel<<<cdiv(N * HC, 256), blk, 0, stream>>>(agg, b2, x_nodes, xb, N * HC, HC);

  // ---- classifier on first 10000 nodes: [10000 x 128] @ Wl^T + bl ---------
  classifier_kernel<<<cdiv(10000 * 5, 256), blk, 0, stream>>>(x_nodes, Wl, bl, (float*)d_out, 10000, 128, 5);
}
